// Seq2Seq_79774722556297
// MI455X (gfx1250) — compile-verified, matrix operand order checked by reading, not run on hardware
//
#include <hip/hip_runtime.h>

// ---------------- problem constants ----------------
#define B_   64
#define T_   128
#define TD_  127     // decoder steps (trg[:, :-1])
#define E_   512
#define H_   512
#define G_   1536    // 3*H
#define V_   10000
#define K_   512

typedef __attribute__((ext_vector_type(16))) __bf16 bf16x16;
typedef __attribute__((ext_vector_type(8)))  __bf16 bf16x8;
typedef __attribute__((ext_vector_type(8)))  float  f32x8;
typedef __attribute__((ext_vector_type(4)))  unsigned int u32x4;
typedef __attribute__((ext_vector_type(8)))  int    i32x8;
typedef __attribute__((ext_vector_type(4)))  int    i32x4;

#if defined(__has_builtin)
#if __has_builtin(__builtin_amdgcn_tensor_load_to_lds) && __has_builtin(__builtin_amdgcn_s_wait_tensorcnt)
#define HAS_TDM 1
#endif
#endif

// ---------------- bf16 helpers (no reliance on __bf16 arithmetic) ----------
__device__ __forceinline__ __bf16 f2bf(float f) {
    union { float f; unsigned u; } v; v.f = f;
    unsigned r = v.u + 0x7FFFu + ((v.u >> 16) & 1u);   // round-to-nearest-even
    unsigned short h = (unsigned short)(r >> 16);
    return __builtin_bit_cast(__bf16, h);
}

// fast reciprocal: single v_rcp_f32 (adequate for GRU gates; operands are
// already bf16-rounded, 1-ULP rcp error is noise) -- avoids the ~10-op IEEE
// divide expansion on the serial recurrence path.
__device__ __forceinline__ float fast_rcp(float x) {
    return __builtin_amdgcn_rcpf(x);
}

// ---------------- WMMA fragment loader -------------------------------------
// 16-bit A-matrix 16x32 layout (ISA 7.12.2): lanes 0-15 hold K {0..7,16..23},
// lanes 16-31 hold K {8..15,24..31}, row = lane&15 (+frag offset).
// B stored as [N,K] row-major uses the identical per-lane gather with row = n.
__device__ __forceinline__ bf16x16 load_frag(const __bf16* __restrict__ base,
                                             int row, int ld, int kb, int lane) {
    int koff = (lane & 16) ? 8 : 0;
    const __bf16* p = base + (size_t)row * ld + kb + koff;
    bf16x8 lo = *(const bf16x8*)p;          // 16B load
    bf16x8 hi = *(const bf16x8*)(p + 16);   // 16B load
    bf16x16 f;
#pragma unroll
    for (int i = 0; i < 8; ++i) { f[i] = lo[i]; f[i + 8] = hi[i]; }
    return f;
}

__device__ __forceinline__ f32x8 zero8() {
    f32x8 z;
#pragma unroll
    for (int e = 0; e < 8; ++e) z[e] = 0.f;
    return z;
}

// ---------------- generic bf16 GEMM: C[M,N] = A[M,K] * W[N,K]^T + bias ------
// Workgroup: 256 thr = 8 waves, tile 128(M) x 128(N); wave tile 64 x 32.
// FC=true: special output indexing into logits [B,128,V] at t+1.
template <bool FC>
__global__ __launch_bounds__(256) void gemm_bf16(const __bf16* __restrict__ A,
                                                 const __bf16* __restrict__ W,
                                                 const float* __restrict__ bias,
                                                 float* __restrict__ C,
                                                 int M, int N) {
    const int lane = threadIdx.x & 31;
    const int wave = threadIdx.x >> 5;
    const int wm = wave & 1, wn = wave >> 1;
    const int m0 = blockIdx.x * 128 + wm * 64;
    const int n0 = blockIdx.y * 128 + wn * 32;

    f32x8 acc[4][2];
#pragma unroll
    for (int i = 0; i < 4; ++i)
#pragma unroll
        for (int j = 0; j < 2; ++j) acc[i][j] = zero8();

    int ar[4], br[2];
#pragma unroll
    for (int i = 0; i < 4; ++i) {
        int r = m0 + i * 16 + (lane & 15);
        ar[i] = r < M ? r : (M - 1);
    }
#pragma unroll
    for (int j = 0; j < 2; ++j) {
        int r = n0 + j * 16 + (lane & 15);
        br[j] = r < N ? r : (N - 1);
    }

    for (int kb = 0; kb < K_; kb += 32) {
        bf16x16 a[4], b[2];
#pragma unroll
        for (int i = 0; i < 4; ++i) a[i] = load_frag(A, ar[i], K_, kb, lane);
#pragma unroll
        for (int j = 0; j < 2; ++j) b[j] = load_frag(W, br[j], K_, kb, lane);
#pragma unroll
        for (int i = 0; i < 4; ++i)
#pragma unroll
            for (int j = 0; j < 2; ++j)
                acc[i][j] = __builtin_amdgcn_wmma_f32_16x16x32_bf16(
                    false, a[i], false, b[j], (short)0, acc[i][j], false, false);
    }

    const int mrow_off = (lane >> 4) << 3;   // C/D layout: lanes>=16 hold M=r+8
#pragma unroll
    for (int i = 0; i < 4; ++i)
#pragma unroll
        for (int j = 0; j < 2; ++j)
#pragma unroll
            for (int r = 0; r < 8; ++r) {
                int m = m0 + i * 16 + r + mrow_off;
                int n = n0 + j * 16 + (lane & 15);
                if (m < M && n < N) {
                    float v = acc[i][j][r] + bias[n];
                    if (FC) {
                        int b = m / TD_, t = m % TD_;
                        C[((size_t)(b * T_ + t + 1)) * V_ + n] = v;
                    } else {
                        C[(size_t)m * N + n] = v;
                    }
                }
            }
}

// ---------------- fused GRU recurrent step ----------------------------------
// grid = 8 blocks (64-col slice of H each), block = 128 thr (4 waves).
// h_prev (64x512 bf16 = 64KB) is staged into padded LDS rows by the Tensor
// Data Mover (one instruction, one wave, TENSORcnt-tracked).  gh = h_prev @
// Whh^T for the r/z/n slices is computed with v_wmma_f32_16x16x32_bf16, then
// the full gate nonlinearity and h-update are fused in the epilogue using
// fast v_exp_f32 / v_rcp_f32.
__global__ __launch_bounds__(128) void gru_step(
    const __bf16* __restrict__ hprev_bf, const float* __restrict__ hprev_f,
    float* __restrict__ hnext_f, __bf16* __restrict__ hnext_bf,
    const float* __restrict__ gi, const float* __restrict__ bhh,
    const __bf16* __restrict__ whh, __bf16* __restrict__ out_bf,
    int t, int T) {
    __shared__ __align__(128) __bf16 sh[64 * 520];   // h_prev staged, padded rows
    const int tid = threadIdx.x;

#ifdef HAS_TDM
    if (tid < 32) {                       // one wave issues the tensor DMA
        unsigned lds_base = (unsigned)(size_t)(&sh[0]);   // low 32b = LDS offset
        unsigned long long ga = (unsigned long long)(size_t)hprev_bf;
        // D# group0: count=1, lds_addr, 57-bit global addr, type=2 ("image")
        u32x4 g0 = { 1u, lds_base,
                     (unsigned)(ga & 0xFFFFFFFFu),
                     ((unsigned)((ga >> 32) & 0x01FFFFFFu)) | 0x80000000u };
        // D# group1: data_size=2B; LDS pad: every 256 DWORDs (interval=7)
        // insert 4 DWORDs (amount=3)  -> rows of 512 data + 8 pad bf16 = 520.
        // 1D tile: tensor_dim0 = tile_dim0 = 32768, stride = 32768.
        i32x8 g1 = { (int)0x07D10000,   // pad_amount=3|pad_interval=7|pad_en|dsize=1
                     (int)0x80000000,   // tensor_dim0 low16 in bits[63:48]
                     (int)0x00010000,   // tensor_dim0 hi=0, tensor_dim1=1
                     (int)0x80000000,   // tile_dim0 = 32768 in bits[127:112]
                     1,                 // tile_dim1 = 1
                     32768,             // tensor_dim0_stride low 32
                     0, 0 };
        i32x4 gz = { 0, 0, 0, 0 };      // groups 2/3 unused (<=2D tensor)
#if __clang_major__ >= 23
        i32x8 gz8 = { 0, 0, 0, 0, 0, 0, 0, 0 };
        __builtin_amdgcn_tensor_load_to_lds(g0, g1, gz, gz, gz8, 0);
#else
        __builtin_amdgcn_tensor_load_to_lds(g0, g1, gz, gz, 0);
#endif
        __builtin_amdgcn_s_wait_tensorcnt(0);
    }
    __syncthreads();
#else
    for (int idx = tid * 8; idx < 64 * 512; idx += 128 * 8) {
        int row = idx >> 9, col = idx & 511;
        *(bf16x8*)(sh + row * 520 + col) = *(const bf16x8*)(hprev_bf + idx);
    }
    __syncthreads();
#endif

    const int lane = tid & 31;
    const int w = tid >> 5;                  // wave id: 16-col sub-slice
    const int g = blockIdx.x;                // 64-col slice of H
    const int ncol = g * 64 + w * 16 + (lane & 15);   // column of H this lane owns

    // Prefetch the gi rows the epilogue will need; overlaps with the K-loop.
    {
        const int b0 = (lane >> 4) << 3;
#pragma unroll
        for (int i = 0; i < 4; ++i) {
            size_t gbase = ((size_t)(i * 16 + b0) * T + t) * (size_t)G_;
            __builtin_prefetch(&gi[gbase + ncol], 0, 0);
            __builtin_prefetch(&gi[gbase + 1024 + ncol], 0, 0);
        }
    }

    f32x8 acc[3][4];
#pragma unroll
    for (int q = 0; q < 3; ++q)
#pragma unroll
        for (int i = 0; i < 4; ++i) acc[q][i] = zero8();

    for (int kb = 0; kb < K_; kb += 32) {
        bf16x16 a[4], b[3];
#pragma unroll
        for (int i = 0; i < 4; ++i)
            a[i] = load_frag(sh, i * 16 + (lane & 15), 520, kb, lane);   // LDS
#pragma unroll
        for (int q = 0; q < 3; ++q)
            b[q] = load_frag(whh, q * 512 + ncol, K_, kb, lane);         // global
#pragma unroll
        for (int q = 0; q < 3; ++q)
#pragma unroll
            for (int i = 0; i < 4; ++i)
                acc[q][i] = __builtin_amdgcn_wmma_f32_16x16x32_bf16(
                    false, a[i], false, b[q], (short)0, acc[q][i], false, false);
    }

    const int mrow_off = (lane >> 4) << 3;
    const float bhr = bhh[ncol];
    const float bhz = bhh[512 + ncol];
    const float bhn = bhh[1024 + ncol];
#pragma unroll
    for (int i = 0; i < 4; ++i)
#pragma unroll
        for (int r = 0; r < 8; ++r) {
            int b = i * 16 + r + mrow_off;                 // batch index
            size_t gbase = ((size_t)b * T + t) * (size_t)G_;
            float ir = gi[gbase + ncol];
            float iz = gi[gbase + 512 + ncol];
            float in_ = gi[gbase + 1024 + ncol];
            float ghr = acc[0][i][r] + bhr;
            float ghz = acc[1][i][r] + bhz;
            float ghn = acc[2][i][r] + bhn;
            float rg = fast_rcp(1.f + __expf(-(ir + ghr)));   // sigmoid
            float zg = fast_rcp(1.f + __expf(-(iz + ghz)));   // sigmoid
            float nx = in_ + rg * ghn;
            nx = fminf(fmaxf(nx, -15.f), 15.f);
            float e2 = __expf(2.f * nx);
            float ng = (e2 - 1.f) * fast_rcp(e2 + 1.f);       // tanh
            float hp = hprev_f[b * 512 + ncol];
            float hn = (1.f - zg) * ng + zg * hp;
            hnext_f[b * 512 + ncol] = hn;
            __bf16 hv = f2bf(hn);
            hnext_bf[b * 512 + ncol] = hv;
            if (out_bf) out_bf[((size_t)b * T + t) * 512 + ncol] = hv;
        }
}

// ---------------- small utility kernels -------------------------------------
__global__ void gather_embed(const int* __restrict__ toks,
                             const float* __restrict__ emb,
                             __bf16* __restrict__ xout, int Tout, int Tin) {
    size_t idx = (size_t)blockIdx.x * 256 + threadIdx.x;
    size_t n = (size_t)B_ * Tout * E_;
    if (idx >= n) return;
    int e = (int)(idx & 511);
    int m = (int)(idx >> 9);
    int b = m / Tout, t = m % Tout;
    int tok = toks[b * Tin + t];
    xout[idx] = f2bf(emb[(size_t)tok * E_ + e]);
}

__global__ void f32_to_bf16(const float* __restrict__ s, __bf16* __restrict__ d, int n) {
    int i = blockIdx.x * 256 + threadIdx.x;
    if (i < n) d[i] = f2bf(s[i]);
}

__global__ void zero_f32(float* p, int n) {
    int i = blockIdx.x * 256 + threadIdx.x;
    if (i < n) p[i] = 0.f;
}

__global__ void zero_bf16(__bf16* p, int n) {
    int i = blockIdx.x * 256 + threadIdx.x;
    if (i < n) p[i] = f2bf(0.f);
}

__global__ void zero_t0(float* __restrict__ out) {   // out[:,0,:] = 0
    int i = blockIdx.x * 256 + threadIdx.x;
    if (i < B_ * V_) out[(size_t)(i / V_) * T_ * V_ + (i % V_)] = 0.f;
}

// ---------------- workspace layout (bytes) ----------------------------------
static const size_t OFF_XE  = 0;                                   // B*T*E bf16
static const size_t OFF_X1  = OFF_XE  + (size_t)B_*T_*E_*2;        // B*T*E bf16
static const size_t OFF_DEC = OFF_X1  + (size_t)B_*T_*E_*2;        // B*T*E bf16
static const size_t OFF_GI  = OFF_DEC + (size_t)B_*T_*E_*2;        // B*T*G f32
static const size_t OFF_H   = OFF_GI  + (size_t)B_*T_*G_*4;        // 2L x 2pp x B*H f32
static const size_t OFF_HB  = OFF_H   + (size_t)4*B_*H_*4;         // 2L x 2pp x B*H bf16
static const size_t OFF_WIE = OFF_HB  + (size_t)4*B_*H_*2;         // enc_Wih bf16
static const size_t OFF_WHE = OFF_WIE + (size_t)2*G_*E_*2;         // enc_Whh bf16
static const size_t OFF_WID = OFF_WHE + (size_t)2*G_*E_*2;         // dec_Wih bf16
static const size_t OFF_WHD = OFF_WID + (size_t)2*G_*E_*2;         // dec_Whh bf16
static const size_t OFF_FCW = OFF_WHD + (size_t)2*G_*E_*2;         // fc_W bf16

extern "C" void kernel_launch(void* const* d_in, const int* in_sizes, int n_in,
                              void* d_out, int out_size, void* d_ws, size_t ws_size,
                              hipStream_t stream) {
    (void)in_sizes; (void)n_in; (void)out_size; (void)ws_size;
    const int*   src     = (const int*)  d_in[0];
    const int*   trg     = (const int*)  d_in[1];
    const float* enc_emb = (const float*)d_in[2];
    const float* enc_Wih = (const float*)d_in[3];
    const float* enc_Whh = (const float*)d_in[4];
    const float* enc_bih = (const float*)d_in[5];
    const float* enc_bhh = (const float*)d_in[6];
    const float* dec_emb = (const float*)d_in[7];
    const float* dec_Wih = (const float*)d_in[8];
    const float* dec_Whh = (const float*)d_in[9];
    const float* dec_bih = (const float*)d_in[10];
    const float* dec_bhh = (const float*)d_in[11];
    const float* fc_W    = (const float*)d_in[12];
    const float* fc_b    = (const float*)d_in[13];
    float* out = (float*)d_out;
    char* ws = (char*)d_ws;

    __bf16* xe   = (__bf16*)(ws + OFF_XE);
    __bf16* x1   = (__bf16*)(ws + OFF_X1);
    __bf16* xdec = (__bf16*)(ws + OFF_DEC);
    float*  gi   = (float*) (ws + OFF_GI);
    float*  hF   = (float*) (ws + OFF_H);
    __bf16* hB   = (__bf16*)(ws + OFF_HB);
    __bf16* wie  = (__bf16*)(ws + OFF_WIE);
    __bf16* whe  = (__bf16*)(ws + OFF_WHE);
    __bf16* wid  = (__bf16*)(ws + OFF_WID);
    __bf16* whd  = (__bf16*)(ws + OFF_WHD);
    __bf16* fcw  = (__bf16*)(ws + OFF_FCW);

    auto cdiv = [](int a, int b) { return (a + b - 1) / b; };

    // 1) one-shot bf16 weight conversion (~46 MB reads)
    f32_to_bf16<<<cdiv(2*G_*E_,256),256,0,stream>>>(enc_Wih, wie, 2*G_*E_);
    f32_to_bf16<<<cdiv(2*G_*E_,256),256,0,stream>>>(enc_Whh, whe, 2*G_*E_);
    f32_to_bf16<<<cdiv(2*G_*E_,256),256,0,stream>>>(dec_Wih, wid, 2*G_*E_);
    f32_to_bf16<<<cdiv(2*G_*E_,256),256,0,stream>>>(dec_Whh, whd, 2*G_*E_);
    f32_to_bf16<<<cdiv(V_*E_,256),256,0,stream>>>(fc_W, fcw, V_*E_);

    // 2) zero hidden state (both layers, both ping-pong buffers)
    zero_f32 <<<cdiv(4*B_*H_,256),256,0,stream>>>(hF, 4*B_*H_);
    zero_bf16<<<cdiv(4*B_*H_,256),256,0,stream>>>(hB, 4*B_*H_);

    int cur[2] = {0, 0};
    auto run_layer = [&](const __bf16* Xin, __bf16* Xout,
                         const __bf16* wih, const __bf16* whh,
                         const float* bih, const float* bhh,
                         int l, int Tt, int Mrows) {
        // hoisted input projection: gi = Xin @ Wih_l^T + bih_l
        gemm_bf16<false><<<dim3(cdiv(Mrows,128), cdiv(G_,128)), 256, 0, stream>>>(
            Xin, wih + (size_t)l * G_ * E_, bih + l * G_, gi, Mrows, G_);
        // serial recurrence, ping-pong h buffers
        for (int t = 0; t < Tt; ++t) {
            int c = cur[l];
            gru_step<<<8, 128, 0, stream>>>(
                hB + (size_t)(l*2 + c)     * B_ * H_,
                hF + (size_t)(l*2 + c)     * B_ * H_,
                hF + (size_t)(l*2 + (c^1)) * B_ * H_,
                hB + (size_t)(l*2 + (c^1)) * B_ * H_,
                gi, bhh + l * G_, whh + (size_t)l * G_ * E_, Xout, t, Tt);
            cur[l] ^= 1;
        }
    };

    // ----- encoder (h starts at 0; 128 steps -> final h back in buffer 0) ---
    gather_embed<<<cdiv(B_*T_*E_,256),256,0,stream>>>(src, enc_emb, xe, T_, T_);
    run_layer(xe, x1,      wie, whe, enc_bih, enc_bhh, 0, T_, B_ * T_);
    run_layer(x1, nullptr, wie, whe, enc_bih, enc_bhh, 1, T_, B_ * T_);

    // ----- decoder (reads encoder's final h from the same buffers) ---------
    gather_embed<<<cdiv(B_*TD_*E_,256),256,0,stream>>>(trg, dec_emb, xe, TD_, T_);
    run_layer(xe, x1,   wid, whd, dec_bih, dec_bhh, 0, TD_, B_ * TD_);
    run_layer(x1, xdec, wid, whd, dec_bih, dec_bhh, 1, TD_, B_ * TD_);

    // ----- output: logits[:,1:,:] = dec_out @ fc_W^T + fc_b; [:,0,:] = 0 ----
    zero_t0<<<cdiv(B_*V_,256),256,0,stream>>>(out);
    gemm_bf16<true><<<dim3(cdiv(B_*TD_,128), cdiv(V_,128)), 256, 0, stream>>>(
        xdec, fcw, fc_b, out, B_ * TD_, V_);
}
